// MultiHeadAttentionWithKVCache_8194797601176
// MI455X (gfx1250) — matrix operand order repaired and checked by reading, not compiled, not run on hardware
//
#include <hip/hip_runtime.h>
#include <stdint.h>

#define BQ 4
#define TQ 1024
#define NH 16
#define DKH 128
#define DMOD 2048
#define NCACHE 4096
#define SFULL 5120
#define SC 32                 // attention key-chunk size
#define NCHUNK (SFULL / SC)   // 160

typedef _Float16 v16h __attribute__((ext_vector_type(16)));
typedef _Float16 v8h  __attribute__((ext_vector_type(8)));
typedef _Float16 v4h  __attribute__((ext_vector_type(4)));
typedef float    v8f  __attribute__((ext_vector_type(8)));
typedef float    f32x4 __attribute__((ext_vector_type(4)));

static __device__ __forceinline__ v16h join16(v8h lo, v8h hi) {
  return __builtin_shufflevector(lo, hi, 0,1,2,3,4,5,6,7,8,9,10,11,12,13,14,15);
}

// ---------------------------------------------------------------------------
// Cache concat: pure byte copy -> use CDNA5 async LDS<->memory DMA path.
// Each lane owns a private 16B LDS slot per tensor; ASYNCcnt-tracked.
// ---------------------------------------------------------------------------
__global__ __launch_bounds__(256)
void copy_cache_kernel(const float* __restrict__ ck, const float* __restrict__ cv,
                       float* __restrict__ Ko, float* __restrict__ Vo) {
  __shared__ float stage[2048];  // 8 KB: 256 lanes x 4 floats x 2 tensors
  typedef __attribute__((address_space(3))) float lds_f;
  uint32_t soffK = (uint32_t)(uintptr_t)(lds_f*)&stage[threadIdx.x * 4];
  uint32_t soffV = soffK + 4096u;

  size_t i = (size_t)blockIdx.x * 256 + threadIdx.x;  // vec4 id, 8,388,608 total
  size_t e = i << 2;
  int d  = (int)(e & 127);
  int s  = (int)((e >> 7) & 4095);
  int bh = (int)(e >> 19);
  size_t dstE = ((size_t)bh * SFULL + s) * DKH + d;

  uint64_t srcK = (uint64_t)(uintptr_t)(ck + e);
  uint64_t srcV = (uint64_t)(uintptr_t)(cv + e);
  uint64_t dK   = (uint64_t)(uintptr_t)(Ko + dstE);
  uint64_t dV   = (uint64_t)(uintptr_t)(Vo + dstE);

  asm volatile("global_load_async_to_lds_b128 %0, %1, off"
               :: "v"(soffK), "v"(srcK) : "memory");
  asm volatile("global_load_async_to_lds_b128 %0, %1, off"
               :: "v"(soffV), "v"(srcV) : "memory");
  asm volatile("s_wait_asynccnt 0x0" ::: "memory");
  asm volatile("global_store_async_from_lds_b128 %0, %1, off"
               :: "v"(dK), "v"(soffK) : "memory");
  asm volatile("global_store_async_from_lds_b128 %0, %1, off"
               :: "v"(dV), "v"(soffV) : "memory");
  asm volatile("s_wait_asynccnt 0x0" ::: "memory");
}

// ---------------------------------------------------------------------------
// Y = X(4096x2048) @ W(2048x2048)^T + bias, f16 WMMA, f32 accum.
// Register double-buffered staging: next K-slab loads are in flight while the
// current slab is consumed by WMMA.
// mode 0: f16 -> outH (B,H,T,Dk) (Q); mode 1: f32 -> outF (B,H,5120,Dk) at
// s=4096+t (K/V); mode 2: f32 row-major (output projection).
// ---------------------------------------------------------------------------
__global__ __launch_bounds__(256)
void gemm_nt_kernel(const float* __restrict__ X, const float* __restrict__ W,
                    const float* __restrict__ bias,
                    float* __restrict__ outF, _Float16* __restrict__ outH, int mode) {
  __shared__ _Float16 Xs[128][40];
  __shared__ _Float16 Wls[128][40];

  const int tid = threadIdx.x;
  const int w = tid >> 5, lane = tid & 31, half = lane >> 4, l16 = lane & 15;
  const int blockM = blockIdx.x * 128, blockN = blockIdx.y * 128;
  const int wm = (w >> 2) * 64, wn = (w & 3) * 32;
  const int srow = tid >> 3;          // staging row base (0..31)
  const int sc4  = (tid & 7) << 2;    // staging col (0..28)

  v8f acc[4][2] = {};
  f32x4 bx[2][4], bw[2][4];

  auto loadTile = [&](int k0, int p) {
#pragma unroll
    for (int i = 0; i < 4; ++i) {
      bx[p][i] = *(const f32x4*)(X + (size_t)(blockM + srow + 32*i) * DMOD + k0 + sc4);
      bw[p][i] = *(const f32x4*)(W + (size_t)(blockN + srow + 32*i) * DMOD + k0 + sc4);
    }
    if (k0 + 64 < DMOD) {   // L2 warm-up two slabs ahead (global_prefetch_b8)
      __builtin_prefetch(X + (size_t)(blockM + srow) * DMOD + k0 + 64, 0, 1);
      __builtin_prefetch(W + (size_t)(blockN + srow) * DMOD + k0 + 64, 0, 1);
    }
  };
  auto storeTile = [&](int p) {
#pragma unroll
    for (int i = 0; i < 4; ++i) {
      v4h xh, wh;
#pragma unroll
      for (int j = 0; j < 4; ++j) { xh[j] = (_Float16)bx[p][i][j]; wh[j] = (_Float16)bw[p][i][j]; }
      *(v4h*)&Xs [srow + 32*i][sc4] = xh;
      *(v4h*)&Wls[srow + 32*i][sc4] = wh;
    }
  };
  auto compute = [&]() {
    v16h af[4];
#pragma unroll
    for (int mt = 0; mt < 4; ++mt) {
      const _Float16* xr = &Xs[wm + mt * 16 + l16][0];
      af[mt] = join16(*(const v8h*)(xr + half * 8), *(const v8h*)(xr + 16 + half * 8));
    }
#pragma unroll
    for (int nt = 0; nt < 2; ++nt) {
      const _Float16* wr = &Wls[wn + nt * 16 + l16][0];
      v16h bf = join16(*(const v8h*)(wr + half * 8), *(const v8h*)(wr + 16 + half * 8));
#pragma unroll
      for (int mt = 0; mt < 4; ++mt)
        acc[mt][nt] = __builtin_amdgcn_wmma_f32_16x16x32_f16(
            false, af[mt], false, bf, (short)0, acc[mt][nt], false, false);
    }
  };

  loadTile(0, 0);
  for (int kk = 0; kk < 64; kk += 2) {
    storeTile(0);
    loadTile((kk + 1) * 32, 1);                 // in flight during compute
    __syncthreads(); compute(); __syncthreads();
    storeTile(1);
    if (kk + 2 < 64) loadTile((kk + 2) * 32, 0);
    __syncthreads(); compute(); __syncthreads();
  }

  // epilogue: bias + scatter (C layout: VGPR r -> row r + 8*half, lane l16 -> col)
#pragma unroll
  for (int nt = 0; nt < 2; ++nt) {
    int n = blockN + wn + nt * 16 + l16;
    float bn = bias[n];
#pragma unroll
    for (int mt = 0; mt < 4; ++mt) {
#pragma unroll
      for (int r = 0; r < 8; ++r) {
        int m = blockM + wm + mt * 16 + r + half * 8;
        float val = acc[mt][nt][r] + bn;
        if (mode == 0) {
          int b = m >> 10, t = m & 1023, hh = n >> 7, d = n & 127;
          outH[((size_t)(b * NH + hh) * TQ + t) * DKH + d] = (_Float16)val;
        } else if (mode == 1) {
          int b = m >> 10, t = m & 1023, hh = n >> 7, d = n & 127;
          outF[((size_t)(b * NH + hh) * SFULL + NCACHE + t) * DKH + d] = val;
        } else {
          outF[(size_t)m * DMOD + n] = val;
        }
      }
    }
  }
}

// ---------------------------------------------------------------------------
// Flash attention: block = one (b,h), 128 q rows; 8 waves x 16 rows.
// 32-key chunks; register double-buffered K/V staging; online softmax.
// ---------------------------------------------------------------------------
__global__ __launch_bounds__(256)
void attn_kernel(const _Float16* __restrict__ Q, const float* __restrict__ K,
                 const float* __restrict__ V, float* __restrict__ O) {
  __shared__ _Float16 Ks [SC][136];        // [s][d], +8 pad
  __shared__ _Float16 Vts[DKH][SC + 8];    // transposed [d][s]
  __shared__ _Float16 Ps [8][16][SC + 8];  // per-wave P tile [m][s]

  const int tid = threadIdx.x;
  const int w = tid >> 5, lane = tid & 31, half = lane >> 4, l16 = lane & 15;
  const int bh = blockIdx.x >> 3, tb = blockIdx.x & 7;
  const int b = bh >> 4, h = bh & 15;
  const int t0 = tb * 128 + w * 16;
  const int sr  = tid >> 5;           // staging row base (0..7)
  const int sc4 = (tid & 31) << 2;    // staging col (0..124)

  // Q fragments kept in registers for the whole kernel
  v16h qf[4];
  const _Float16* qrow = Q + ((size_t)bh * TQ + t0 + l16) * DKH;
#pragma unroll
  for (int ks = 0; ks < 4; ++ks)
    qf[ks] = join16(*(const v8h*)(qrow + ks * 32 + half * 8),
                    *(const v8h*)(qrow + ks * 32 + 16 + half * 8));

  v8f oacc[8] = {};
  float rmax[8], rsum[8];
#pragma unroll
  for (int r = 0; r < 8; ++r) { rmax[r] = -1e30f; rsum[r] = 0.f; }
  const float scale = 0.08838834764831845f;  // 1/sqrt(128)

  f32x4 kx[2][4], vx[2][4];
  auto loadChunk = [&](int c, int p) {
#pragma unroll
    for (int i = 0; i < 4; ++i) {
      size_t g = ((size_t)bh * SFULL + c * SC + sr + 8*i) * DKH + sc4;
      kx[p][i] = *(const f32x4*)(K + g);
      vx[p][i] = *(const f32x4*)(V + g);
    }
  };
  auto storeChunk = [&](int p) {
#pragma unroll
    for (int i = 0; i < 4; ++i) {
      int row = sr + 8*i;
      v4h kh;
#pragma unroll
      for (int j = 0; j < 4; ++j) kh[j] = (_Float16)kx[p][i][j];
      *(v4h*)&Ks[row][sc4] = kh;
      Vts[sc4 + 0][row] = (_Float16)vx[p][i][0];
      Vts[sc4 + 1][row] = (_Float16)vx[p][i][1];
      Vts[sc4 + 2][row] = (_Float16)vx[p][i][2];
      Vts[sc4 + 3][row] = (_Float16)vx[p][i][3];
    }
  };
  auto computeChunk = [&]() {
    // scores: 16x32 = 2 n-tiles, each over Dk with 4 WMMAs
    v8f sc[2];
#pragma unroll
    for (int nt = 0; nt < 2; ++nt) {
      v8f s = {};
      const _Float16* kr = &Ks[nt * 16 + l16][0];
#pragma unroll
      for (int ks = 0; ks < 4; ++ks) {
        v16h bf = join16(*(const v8h*)(kr + ks * 32 + half * 8),
                         *(const v8h*)(kr + ks * 32 + 16 + half * 8));
        s = __builtin_amdgcn_wmma_f32_16x16x32_f16(false, qf[ks], false, bf,
                                                   (short)0, s, false, false);
      }
#pragma unroll
      for (int r = 0; r < 8; ++r) s[r] *= scale;
      sc[nt] = s;
    }
    // online softmax (row r lives in 16-lane halves -> xor masks 1,2,4,8)
#pragma unroll
    for (int r = 0; r < 8; ++r) {
      float m_ = fmaxf(sc[0][r], sc[1][r]);
      m_ = fmaxf(m_, __shfl_xor(m_, 1, 32));
      m_ = fmaxf(m_, __shfl_xor(m_, 2, 32));
      m_ = fmaxf(m_, __shfl_xor(m_, 4, 32));
      m_ = fmaxf(m_, __shfl_xor(m_, 8, 32));
      float mnew = fmaxf(rmax[r], m_);
      float corr = __expf(rmax[r] - mnew);
      rmax[r] = mnew;
      float p0 = __expf(sc[0][r] - mnew);
      float p1 = __expf(sc[1][r] - mnew);
      sc[0][r] = p0; sc[1][r] = p1;
      float ssum = p0 + p1;
      ssum += __shfl_xor(ssum, 1, 32);
      ssum += __shfl_xor(ssum, 2, 32);
      ssum += __shfl_xor(ssum, 4, 32);
      ssum += __shfl_xor(ssum, 8, 32);
      rsum[r] = rsum[r] * corr + ssum;
#pragma unroll
      for (int dt = 0; dt < 8; ++dt) oacc[dt][r] *= corr;
    }
    // C layout -> A layout via per-wave LDS round trip
#pragma unroll
    for (int nt = 0; nt < 2; ++nt)
#pragma unroll
      for (int r = 0; r < 8; ++r)
        Ps[w][r + half * 8][nt * 16 + l16] = (_Float16)sc[nt][r];
    asm volatile("s_wait_dscnt 0x0" ::: "memory");  // own-wave DS RAW fence
    // O += P @ V (single K=32 step)
    const _Float16* pr = &Ps[w][l16][0];
    v16h pf = join16(*(const v8h*)(pr + half * 8), *(const v8h*)(pr + 16 + half * 8));
#pragma unroll
    for (int dt = 0; dt < 8; ++dt) {
      const _Float16* vr = &Vts[dt * 16 + l16][0];
      v16h bf = join16(*(const v8h*)(vr + half * 8), *(const v8h*)(vr + 16 + half * 8));
      oacc[dt] = __builtin_amdgcn_wmma_f32_16x16x32_f16(
          false, pf, false, bf, (short)0, oacc[dt], false, false);
    }
  };

  loadChunk(0, 0);
  for (int c = 0; c < NCHUNK; c += 2) {
    storeChunk(0);
    loadChunk(c + 1, 1);                        // in flight during compute
    __syncthreads(); computeChunk(); __syncthreads();
    storeChunk(1);
    if (c + 2 < NCHUNK) loadChunk(c + 2, 0);
    __syncthreads(); computeChunk(); __syncthreads();
  }

  // normalize, write (B,T,D_MODEL) for the Wo GEMM
#pragma unroll
  for (int dt = 0; dt < 8; ++dt) {
#pragma unroll
    for (int r = 0; r < 8; ++r) {
      int t = t0 + r + half * 8;
      int d = dt * 16 + l16;
      O[((size_t)b * TQ + t) * DMOD + h * DKH + d] = oacc[dt][r] / rsum[r];
    }
  }
}

// ---------------------------------------------------------------------------
extern "C" void kernel_launch(void* const* d_in, const int* in_sizes, int n_in,
                              void* d_out, int out_size, void* d_ws, size_t ws_size,
                              hipStream_t stream) {
  (void)in_sizes; (void)n_in; (void)out_size; (void)ws_size;
  const float* x    = (const float*)d_in[0];
  const float* ck   = (const float*)d_in[1];
  const float* cv   = (const float*)d_in[2];
  const float* Wq_w = (const float*)d_in[3];
  const float* Wq_b = (const float*)d_in[4];
  const float* Wk_w = (const float*)d_in[5];
  const float* Wk_b = (const float*)d_in[6];
  const float* Wv_w = (const float*)d_in[7];
  const float* Wv_b = (const float*)d_in[8];
  const float* Wo_w = (const float*)d_in[9];
  const float* Wo_b = (const float*)d_in[10];

  float* out  = (float*)d_out;
  float* outO = out;                                   // (B,T,D)
  float* outK = out + (size_t)BQ * TQ * DMOD;          // (B,H,S,Dk)
  float* outV = outK + (size_t)BQ * NH * SFULL * DKH;  // (B,H,S,Dk)

  _Float16* qws = (_Float16*)d_ws;                                         // 16 MB
  float* attnws = (float*)((char*)d_ws + (size_t)BQ * NH * TQ * DKH * 2);  // 32 MB

  copy_cache_kernel<<<32768, 256, 0, stream>>>(ck, cv, outK, outV);

  dim3 g(32, 16, 1);
  gemm_nt_kernel<<<g, 256, 0, stream>>>(x, Wq_w, Wq_b, nullptr, qws, 0);
  gemm_nt_kernel<<<g, 256, 0, stream>>>(x, Wk_w, Wk_b, outK, nullptr, 1);
  gemm_nt_kernel<<<g, 256, 0, stream>>>(x, Wv_w, Wv_b, outV, nullptr, 1);

  attn_kernel<<<512, 256, 0, stream>>>(qws, outK, outV, attnws);

  gemm_nt_kernel<<<g, 256, 0, stream>>>(attnws, Wo_w, Wo_b, outO, nullptr, 2);
}
